// EncoderDecoder_3470333575658
// MI455X (gfx1250) — matrix operand order repaired and checked by reading, not compiled, not run on hardware
//
#include <hip/hip_runtime.h>
#include <hip/hip_bf16.h>

// Problem constants (from reference)
#define Bb 64
#define Aa 23
#define Tt 128
#define Dd 256
#define AE 32
#define Ss 6
#define Hh 8
#define DK 32
#define DM 288            // D + AE
#define BLOCK 256         // 8 wave32 waves
#define MSTR 296          // mem_h row stride (halfs), 16B-aligned rows
#define TSTR 136          // mem_t row stride (halfs), 16B-aligned rows
#define SCSTR 17          // scores row stride (floats), odd -> conflict-free col reads

typedef __attribute__((ext_vector_type(16))) _Float16 v16h;
typedef __attribute__((ext_vector_type(8)))  _Float16 v8h;
typedef __attribute__((ext_vector_type(8)))  float    v8f;

// ---- dynamic LDS layout (bytes), all offsets 16B-aligned ----
#define OFF_MEMH 0                        // _Float16[128*296]  75776 B (t-major)
#define OFF_MEMT 75776                    // _Float16[288*136]  78336 B (e-major)
#define OFF_SC   154112                   // float[128*17]       8704 B
#define OFF_PF   162816                   // _Float16[4*32*16]   4096 B (A-frag layout)
#define OFF_PM   166912                   // float[8*288]        9216 B
#define OFF_QKF  176128                   // _Float16[9*32*16]   9216 B (B-frag layout)
#define OFF_Q    185344                   // float[256]          1024 B
#define OFF_AW   186368                   // float[288]          1152 B
#define OFF_CTX  187520                   // float[256]          1024 B
#define SMEM_BYTES 188544

__global__ void zero_ws_kernel(float* ws, int n) {
    int i = blockIdx.x * blockDim.x + threadIdx.x;
    if (i < n) ws[i] = 0.0f;
}

// One block per (b,a): builds masked memory tile in LDS (f16, both
// orientations), pools it, folds q into Wk (Qk), runs two WMMA GEMMs
// (scores = mem @ Qk, pm = p @ mem), projects through Wv/Wo, and atomically
// accumulates the agent mean. All WMMA operands are contiguous b128 LDS loads.
__global__ void __launch_bounds__(BLOCK)
agent_kernel(const float* __restrict__ data,      // (B,A,T)
             const int*   __restrict__ timei,     // (B,A,T)
             const int*   __restrict__ lengths,   // (B,A)
             const float* __restrict__ W_embed,   // (A,D)
             const float* __restrict__ b_embed,   // (A,D)
             const float* __restrict__ E_var,     // (A,AE)
             const float* __restrict__ Wq,        // (DM,D)
             const float* __restrict__ bq,        // (D)
             const float* __restrict__ Wk,        // (DM,D)
             const float* __restrict__ Wv,        // (DM,D)
             const float* __restrict__ bv,        // (D)
             const float* __restrict__ Wo,        // (D,DM)
             const float* __restrict__ bo,        // (DM)
             float* __restrict__ aggre,           // (B,DM) atomic accum
             float* __restrict__ nagents)         // (B)    atomic accum
{
    extern __shared__ char smem[];
    _Float16* mem_h = (_Float16*)(smem + OFF_MEMH);
    _Float16* mem_t = (_Float16*)(smem + OFF_MEMT);
    float*    sc    = (float*)   (smem + OFF_SC);
    _Float16* p_f   = (_Float16*)(smem + OFF_PF);
    float*    pm    = (float*)   (smem + OFF_PM);
    _Float16* qk_f  = (_Float16*)(smem + OFF_QKF);
    float*    qv    = (float*)   (smem + OFF_Q);
    float*    aw    = (float*)   (smem + OFF_AW);
    float*    ctx   = (float*)   (smem + OFF_CTX);
    __shared__ int s_cnt;

    const int bId = blockIdx.x / Aa;
    const int aId = blockIdx.x % Aa;
    const int tid  = threadIdx.x;
    const int lane = tid & 31;
    const int wid  = tid >> 5;

    const float* data_row = data  + ((size_t)bId * Aa + aId) * Tt;
    const int*   time_row = timei + ((size_t)bId * Aa + aId) * Tt;
    const float* We = W_embed + (size_t)aId * Dd;
    const float* be = b_embed + (size_t)aId * Dd;
    const float* Ev = E_var   + (size_t)aId * AE;

    if (tid == 0) s_cnt = 0;
    __syncthreads();

    // ---- Phase 1: build masked memory tile (f16) in BOTH orientations ----
    // memory[t][e<256]  = relu(data[t]*We[e]+be[e]) + PE(t,e)   if time>=0
    // memory[t][e>=256] = E_var[e-256]                          if time>=0
    const float PE_FACT = 0.07195578415606392f; // ln(10000)*2/256
    for (int idx = tid; idx < Tt * DM; idx += BLOCK) {
        int t = idx / DM;
        int e = idx - t * DM;
        int tm = time_row[t];
        float val = 0.0f;
        if (tm >= 0) {
            if (e < Dd) {
                float x = fmaf(data_row[t], We[e], be[e]);
                x = fmaxf(x, 0.0f);
                float ang = (float)tm * __expf(-(float)(e >> 1) * PE_FACT);
                float pe = (e & 1) ? __cosf(ang) : __sinf(ang);
                val = x + pe;
            } else {
                val = Ev[e - Dd];
            }
        }
        _Float16 hv = (_Float16)val;
        mem_h[t * MSTR + e] = hv;     // t-major: feeds scores A-fragments
        mem_t[e * TSTR + t] = hv;     // e-major: feeds pm B-fragments + pooling
    }
    if (tid < Tt && time_row[tid] >= 0) atomicAdd(&s_cnt, 1);
    __syncthreads();

    // ---- Phase 1b: agentwise mean over T (contiguous mem_t rows) ---------
    const float cntf = (float)s_cnt;
    for (int e = tid; e < DM; e += BLOCK) {
        float s = 0.0f;
        for (int t = 0; t < Tt; ++t) s += (float)mem_t[e * TSTR + t];
        aw[e] = s / (cntf + 1e-9f);
    }
    __syncthreads();

    // ---- Phase 2: q = agentwise @ Wq + bq --------------------------------
    {
        int d = tid;              // BLOCK == D
        float s = bq[d];
        for (int e = 0; e < DM; ++e) s = fmaf(aw[e], Wq[(size_t)e * Dd + d], s);
        qv[d] = s;
    }
    __syncthreads();

    // ---- Phase 3: fold q into Wk, write DIRECTLY in B-fragment layout ----
    // B-frag (16x16x32, K-step ks): lane = n + 16*(r>=16), elem = r&15, r=K&31
    for (int i = tid; i < DM * Hh; i += BLOCK) {
        int e = i >> 3;
        int h = i & 7;
        const float* wk = Wk + (size_t)e * Dd + h * DK;
        const float* qh = qv + h * DK;
        float s = 0.0f;
        #pragma unroll
        for (int d2 = 0; d2 < DK; ++d2) s = fmaf(wk[d2], qh[d2], s);
        int ks = e >> 5;
        int r  = e & 31;
        int fl = h + 16 * (r >> 4);
        int el = r & 15;
        qk_f[(ks * 32 + fl) * 16 + el]       = (_Float16)s;
        qk_f[(ks * 32 + 8 + fl) * 16 + el]   = (_Float16)0.0f;  // pad cols 8..15
    }
    __syncthreads();

    // ---- Phase 4: scores = memory(128x288) @ Qk(288x16) via WMMA ---------
    {
        v8f acc = {};
        const int mrow = 16 * wid + (lane & 15);   // wave handles M-tile = wid
        const int kba  = (lane >> 4) * 16;
        #pragma unroll
        for (int ks = 0; ks < 9; ++ks) {           // K = 288 = 9*32
            const _Float16* rp = mem_h + mrow * MSTR + ks * 32 + kba;
            v8h a0 = *(const v8h*)rp;              // K = k0+kba .. +7
            v8h a1 = *(const v8h*)(rp + 8);        // K = k0+kba+8 .. +15
            v16h af = __builtin_shufflevector(a0, a1,
                0,1,2,3,4,5,6,7,8,9,10,11,12,13,14,15);
            const _Float16* bp = qk_f + (ks * 32 + lane) * 16;
            v8h b0 = *(const v8h*)bp;
            v8h b1 = *(const v8h*)(bp + 8);
            v16h bf = __builtin_shufflevector(b0, b1,
                0,1,2,3,4,5,6,7,8,9,10,11,12,13,14,15);
            acc = __builtin_amdgcn_wmma_f32_16x16x32_f16(
                false, af, false, bf, (short)0, acc, false, false);
        }
        const int bcol = lane & 15;
        #pragma unroll
        for (int j = 0; j < 8; ++j) {
            int m = j + 8 * (lane >> 4);
            sc[(16 * wid + m) * SCSTR + bcol] = acc[j];
        }
    }
    __syncthreads();

    // ---- Phase 5: masked softmax over T per head; write p in A-frag layout
    // A-frag (K-step ks): lane = h + 16*((r>>3)&1), elem = (r&7)+8*(r>>4)
    {
        const int h   = wid;                       // 8 waves == 8 heads
        const int len = lengths[(size_t)bId * Aa + aId];
        float sv[4];
        float mx = -3.0e38f;
        #pragma unroll
        for (int i = 0; i < 4; ++i) {
            int t = lane + 32 * i;
            float s = sc[t * SCSTR + h] * 0.17677669529663687f;  // 1/sqrt(32)
            bool am = (len > 0) && (t < len);       // amask = valid[0]&valid[t]
            sv[i] = am ? s : -1.0e10f;
            mx = fmaxf(mx, sv[i]);
        }
        for (int off = 16; off >= 1; off >>= 1) mx = fmaxf(mx, __shfl_xor(mx, off, 32));
        float se = 0.0f;
        #pragma unroll
        for (int i = 0; i < 4; ++i) { sv[i] = __expf(sv[i] - mx); se += sv[i]; }
        for (int off = 16; off >= 1; off >>= 1) se += __shfl_xor(se, off, 32);
        float inv = 1.0f / se;
        const int fl = 16 * ((lane >> 3) & 1);          // r == lane here
        const int el = (lane & 7) + 8 * (lane >> 4);
        #pragma unroll
        for (int i = 0; i < 4; ++i) {                   // ks == i
            p_f[(i * 32 + h + fl) * 16 + el]     = (_Float16)(sv[i] * inv);
            p_f[(i * 32 + h + 8 + fl) * 16 + el] = (_Float16)0.0f;  // pad M 8..15
        }
    }
    __syncthreads();

    // ---- Phase 6: pm = p(16x128) @ memory(128x288) via WMMA --------------
    for (int nt = wid; nt < DM / 16; nt += 8) {    // 18 N-tiles over 8 waves
        v8f acc = {};
        const int ecol = nt * 16 + (lane & 15);
        const int kbb  = (lane >> 4) * 8;
        #pragma unroll
        for (int ks = 0; ks < 4; ++ks) {           // K = 128 = 4*32
            const _Float16* ap = p_f + (ks * 32 + lane) * 16;
            v8h a0 = *(const v8h*)ap;
            v8h a1 = *(const v8h*)(ap + 8);
            v16h af = __builtin_shufflevector(a0, a1,
                0,1,2,3,4,5,6,7,8,9,10,11,12,13,14,15);
            const _Float16* bp = mem_t + ecol * TSTR + ks * 32 + kbb;
            v8h b0 = *(const v8h*)bp;              // K = k0+kbb .. +7
            v8h b1 = *(const v8h*)(bp + 16);       // K = k0+16+kbb .. +7
            v16h bf = __builtin_shufflevector(b0, b1,
                0,1,2,3,4,5,6,7,8,9,10,11,12,13,14,15);
            acc = __builtin_amdgcn_wmma_f32_16x16x32_f16(
                false, af, false, bf, (short)0, acc, false, false);
        }
        if (lane < 16) {                           // rows 0..7 are real heads
            #pragma unroll
            for (int j = 0; j < 8; ++j) pm[j * DM + ecol] = acc[j];
        }
    }
    __syncthreads();

    // ---- Phase 7: ctx[d] = bv[d] + pm[h] . Wv[:, d]   (softmax sums to 1)
    {
        int d = tid;
        int h = d >> 5;
        const float* pmh = pm + h * DM;
        float s = bv[d];
        for (int e = 0; e < DM; ++e) s = fmaf(pmh[e], Wv[(size_t)e * Dd + d], s);
        ctx[d] = s;
    }
    __syncthreads();

    // ---- Phase 8: agent_feat = ctx @ Wo + bo, masked mean over agents ----
    const bool agent_ok = (s_cnt > 0);
    for (int o = tid; o < DM; o += BLOCK) {
        float s = bo[o];
        for (int d = 0; d < Dd; ++d) s = fmaf(ctx[d], Wo[(size_t)d * DM + o], s);
        if (agent_ok) atomicAdd(&aggre[(size_t)bId * DM + o], s);
    }
    if (tid == 0 && agent_ok) atomicAdd(&nagents[bId], 1.0f);
}

// One block per batch element: aggre mean, static projection, final dot.
__global__ void __launch_bounds__(256)
final_kernel(const float* __restrict__ aggre, const float* __restrict__ nagents,
             const float* __restrict__ statq, const float* __restrict__ W_static,
             const float* __restrict__ b_static, const float* __restrict__ Wg,
             const float* __restrict__ bg, float* __restrict__ out)
{
    __shared__ float red[256];
    const int b = blockIdx.x;
    const int tid = threadIdx.x;
    const float denom = nagents[b] + 1e-9f;
    float part = 0.0f;
    for (int o = tid; o < DM; o += 256)
        part += (aggre[(size_t)b * DM + o] / denom) * Wg[o];
    {
        int d = tid;   // 256 threads == D
        float sf = b_static[d];
        #pragma unroll
        for (int s = 0; s < Ss; ++s) sf = fmaf(statq[b * Ss + s], W_static[s * Dd + d], sf);
        part += sf * Wg[DM + d];
    }
    red[tid] = part;
    __syncthreads();
    for (int off = 128; off >= 1; off >>= 1) {
        if (tid < off) red[tid] += red[tid + off];
        __syncthreads();
    }
    if (tid == 0) out[b] = red[0] + bg[0];
}

extern "C" void kernel_launch(void* const* d_in, const int* in_sizes, int n_in,
                              void* d_out, int out_size, void* d_ws, size_t ws_size,
                              hipStream_t stream) {
    // setup_inputs() order:
    const float* data     = (const float*)d_in[0];
    const int*   timei    = (const int*)  d_in[1];
    const int*   lengths  = (const int*)  d_in[2];
    const float* statq    = (const float*)d_in[3];
    const float* W_embed  = (const float*)d_in[4];
    const float* b_embed  = (const float*)d_in[5];
    const float* E_var    = (const float*)d_in[6];
    const float* W_static = (const float*)d_in[7];
    const float* b_static = (const float*)d_in[8];
    const float* Wq       = (const float*)d_in[9];
    const float* bq       = (const float*)d_in[10];
    const float* Wk       = (const float*)d_in[11];
    // d_in[12] = bk: provably cancels in softmax (constant per head over t)
    const float* Wv       = (const float*)d_in[13];
    const float* bv       = (const float*)d_in[14];
    const float* Wo       = (const float*)d_in[15];
    const float* bo       = (const float*)d_in[16];
    const float* Wg       = (const float*)d_in[17];
    const float* bg       = (const float*)d_in[18];
    float* out = (float*)d_out;

    float* aggre   = (float*)d_ws;            // B*DM floats
    float* nagents = aggre + Bb * DM;         // B floats
    const int nzero = Bb * DM + Bb;

    (void)hipFuncSetAttribute((const void*)agent_kernel,
                              hipFuncAttributeMaxDynamicSharedMemorySize,
                              SMEM_BYTES);

    zero_ws_kernel<<<(nzero + 255) / 256, 256, 0, stream>>>(aggre, nzero);

    agent_kernel<<<Bb * Aa, BLOCK, SMEM_BYTES, stream>>>(
        data, timei, lengths, W_embed, b_embed, E_var,
        Wq, bq, Wk, Wv, bv, Wo, bo, aggre, nagents);

    final_kernel<<<Bb, 256, 0, stream>>>(
        aggre, nagents, statq, W_static, b_static, Wg, bg, out);
}
